// VectorQuantizer_41197326303601
// MI455X (gfx1250) — matrix operand order repaired and checked by reading, not compiled, not run on hardware
//
#include <hip/hip_runtime.h>
#include <math.h>

typedef __attribute__((ext_vector_type(16))) __bf16 v16bf;
typedef __attribute__((ext_vector_type(8)))  float  v8f;

#define DDIM 64
#define EPSN 1e-8f
#define NEG_BIG (-1.0e9f)

// ---------- Kernel 1: x_n = x/(||x||+eps); emit f32 to d_out and bf16 hi/lo ----------
__global__ void vq_norm_x(const float* __restrict__ x, float* __restrict__ xn_out,
                          __bf16* __restrict__ x_hi, __bf16* __restrict__ x_lo, int B) {
  int row  = (int)((blockIdx.x * blockDim.x + threadIdx.x) >> 5);
  int lane = threadIdx.x & 31;
  if (row >= B) return;
  const float* xr = x + (size_t)row * DDIM;
  float e0 = xr[lane], e1 = xr[lane + 32];
  float ss = e0 * e0 + e1 * e1;
  #pragma unroll
  for (int m = 16; m >= 1; m >>= 1) ss += __shfl_xor(ss, m, 32);
  float inv = 1.0f / (sqrtf(ss) + EPSN);
  float n0 = e0 * inv, n1 = e1 * inv;
  size_t o = (size_t)row * DDIM;
  xn_out[o + lane] = n0;  xn_out[o + 32 + lane] = n1;
  __bf16 h0 = (__bf16)n0, h1 = (__bf16)n1;
  x_hi[o + lane] = h0;                       x_hi[o + 32 + lane] = h1;
  x_lo[o + lane] = (__bf16)(n0 - (float)h0); x_lo[o + 32 + lane] = (__bf16)(n1 - (float)h1);
}

// ---------- Kernel 2: cb = normalize(codebook); cb_n = normalize(cb) ----------
__global__ void vq_norm_cb(const float* __restrict__ cbi, float* __restrict__ cb_f32,
                           __bf16* __restrict__ cb_hi, __bf16* __restrict__ cb_lo, int K) {
  int row  = (int)((blockIdx.x * blockDim.x + threadIdx.x) >> 5);
  int lane = threadIdx.x & 31;
  if (row >= K) return;
  const float* cr = cbi + (size_t)row * DDIM;
  float e0 = cr[lane], e1 = cr[lane + 32];
  float ss = e0 * e0 + e1 * e1;
  #pragma unroll
  for (int m = 16; m >= 1; m >>= 1) ss += __shfl_xor(ss, m, 32);
  float inv0 = 1.0f / (sqrtf(ss) + EPSN);
  float c0 = e0 * inv0, c1 = e1 * inv0;       // once-normalized (used for gather z)
  size_t o = (size_t)row * DDIM;
  cb_f32[o + lane] = c0;  cb_f32[o + 32 + lane] = c1;
  float ss2 = c0 * c0 + c1 * c1;
  #pragma unroll
  for (int m = 16; m >= 1; m >>= 1) ss2 += __shfl_xor(ss2, m, 32);
  float inv1 = 1.0f / (sqrtf(ss2) + EPSN);
  float n0 = c0 * inv1, n1 = c1 * inv1;       // twice-normalized (used for distances)
  __bf16 h0 = (__bf16)n0, h1 = (__bf16)n1;
  cb_hi[o + lane] = h0;                       cb_hi[o + 32 + lane] = h1;
  cb_lo[o + lane] = (__bf16)(n0 - (float)h0); cb_lo[o + 32 + lane] = (__bf16)(n1 - (float)h1);
}

// ---------- Kernel 3: pack mask[B][K] (bool bytes) -> packM[B/16][K] u16 bit-tiles ----------
__global__ void vq_pack_mask(const unsigned char* __restrict__ mask,
                             unsigned short* __restrict__ packM, int B, int K) {
  size_t gid = (size_t)blockIdx.x * blockDim.x + threadIdx.x;
  size_t total = (size_t)(B >> 4) * (size_t)K;
  if (gid >= total) return;
  int    j  = (int)(gid % (size_t)K);
  size_t rt = gid / (size_t)K;
  const unsigned char* mp = mask + rt * 16u * (size_t)K + (size_t)j;
  unsigned w = 0;
  #pragma unroll
  for (int r = 0; r < 16; ++r) w |= (unsigned)(mp[(size_t)r * K] != 0) << r;
  packM[gid] = (unsigned short)w;
}

// One 16x16 distance tile via bf16x3 split: two independent 3-WMMA chains, merged by VALU.
__device__ __forceinline__ v8f vq_tile_acc(const v16bf& a0h, const v16bf& a0l,
                                           const v16bf& a1h, const v16bf& a1l,
                                           const __bf16* __restrict__ cb_hi,
                                           const __bf16* __restrict__ cb_lo, size_t bb) {
  v16bf b0h = *(const v16bf*)(cb_hi + bb);
  v16bf b1h = *(const v16bf*)(cb_hi + bb + 32);
  v16bf b0l = *(const v16bf*)(cb_lo + bb);
  v16bf b1l = *(const v16bf*)(cb_lo + bb + 32);
  v8f p = {};
  v8f q = {};
  p = __builtin_amdgcn_wmma_f32_16x16x32_bf16(false, a0h, false, b0h, (short)0, p, false, false);
  q = __builtin_amdgcn_wmma_f32_16x16x32_bf16(false, a1l, false, b1h, (short)0, q, false, false);
  p = __builtin_amdgcn_wmma_f32_16x16x32_bf16(false, a1h, false, b1h, (short)0, p, false, false);
  q = __builtin_amdgcn_wmma_f32_16x16x32_bf16(false, a0h, false, b0l, (short)0, q, false, false);
  p = __builtin_amdgcn_wmma_f32_16x16x32_bf16(false, a0l, false, b0h, (short)0, p, false, false);
  q = __builtin_amdgcn_wmma_f32_16x16x32_bf16(false, a1h, false, b1l, (short)0, q, false, false);
  return p + q;
}

// ---------- Kernel 4: fused bf16x3 WMMA GEMM + masked argmin. 1 wave = 16 rows ----------
__global__ void __launch_bounds__(256)
vq_argmin_wmma(const __bf16* __restrict__ x_hi, const __bf16* __restrict__ x_lo,
               const __bf16* __restrict__ cb_hi, const __bf16* __restrict__ cb_lo,
               const unsigned short* __restrict__ packM, const int* __restrict__ training,
               int* __restrict__ idx_out, int B, int K) {
  int lane = threadIdx.x & 31;
  int rt   = blockIdx.x * (blockDim.x >> 5) + (threadIdx.x >> 5);
  if (rt >= (B >> 4)) return;
  int h = lane >> 4;        // wave half
  int m = lane & 15;        // A row within tile / B column class
  unsigned allmask = (*training != 0) ? 0u : 0xFFFFu;

  // A fragments (x rows rt*16+m, bf16 hi/lo), 16x32 layout: K=16h+i
  const __bf16* xh = x_hi + (size_t)(rt * 16 + m) * DDIM;
  const __bf16* xl = x_lo + (size_t)(rt * 16 + m) * DDIM;
  v16bf a0h, a0l, a1h, a1l;
  #pragma unroll
  for (int i = 0; i < 16; ++i) {
    int k = 16 * h + i;
    a0h[i] = xh[k];      a0l[i] = xl[k];
    a1h[i] = xh[32 + k]; a1l[i] = xl[32 + k];
  }

  float best[8];
  int   bidx[8];
  #pragma unroll
  for (int v = 0; v < 8; ++v) { best[v] = -INFINITY; bidx[v] = 0; }

  const unsigned short* mrow = packM + (size_t)rt * (size_t)K;
  int nTiles = K >> 4;
  int sh = 8 * h;

  // two column tiles per iteration -> 12 WMMAs across 4 independent chains
  int t = 0;
  for (; t + 1 < nTiles; t += 2) {
    int j0 = t << 4;
    size_t bbA = (size_t)(j0 + m) * DDIM + (size_t)(16 * h);
    size_t bbB = bbA + (size_t)16 * DDIM;
    unsigned mwA = (unsigned)mrow[j0 + m]      | allmask;
    unsigned mwB = (unsigned)mrow[j0 + 16 + m] | allmask;
    if ((t & 7) == 0) __builtin_prefetch(&mrow[j0 + 256], 0, 0);

    v8f accA = vq_tile_acc(a0h, a0l, a1h, a1l, cb_hi, cb_lo, bbA);
    v8f accB = vq_tile_acc(a0h, a0l, a1h, a1l, cb_hi, cb_lo, bbB);

    int jA = j0 + m, jB = j0 + 16 + m;
    #pragma unroll
    for (int v = 0; v < 8; ++v) {            // row M = v + 8h, col N = m
      float sA = ((mwA >> (v + sh)) & 1u) ? accA[v] : NEG_BIG;
      if (sA > best[v]) { best[v] = sA; bidx[v] = jA; }  // strict > keeps first occurrence
      float sB = ((mwB >> (v + sh)) & 1u) ? accB[v] : NEG_BIG;
      if (sB > best[v]) { best[v] = sB; bidx[v] = jB; }
    }
  }
  if (t < nTiles) {                          // odd tile-count epilogue
    int j0 = t << 4;
    size_t bb = (size_t)(j0 + m) * DDIM + (size_t)(16 * h);
    unsigned mw = (unsigned)mrow[j0 + m] | allmask;
    v8f acc = vq_tile_acc(a0h, a0l, a1h, a1l, cb_hi, cb_lo, bb);
    int j = j0 + m;
    #pragma unroll
    for (int v = 0; v < 8; ++v) {
      float s = ((mw >> (v + sh)) & 1u) ? acc[v] : NEG_BIG;
      if (s > best[v]) { best[v] = s; bidx[v] = j; }
    }
  }

  // reduce across the 16 column classes (xor<=8 stays inside each 16-lane half)
  #pragma unroll
  for (int v = 0; v < 8; ++v) {
    float bv = best[v]; int bi = bidx[v];
    #pragma unroll
    for (int off = 8; off >= 1; off >>= 1) {
      float ov = __shfl_xor(bv, off, 32);
      int   oi = __shfl_xor(bi, off, 32);
      if (ov > bv || (ov == bv && oi < bi)) { bv = ov; bi = oi; }
    }
    best[v] = bv; bidx[v] = bi;
  }
  if (m == 0) {
    #pragma unroll
    for (int v = 0; v < 8; ++v) idx_out[rt * 16 + 8 * h + v] = bidx[v];
  }
}

// ---------- Kernel 5: gather z = cb[idx]; z_q = x_n + (z - x_n); idx -> float ----------
__global__ void vq_gather(const int* __restrict__ idx, const float* __restrict__ cb_f32,
                          const float* __restrict__ xn, float* __restrict__ zq,
                          float* __restrict__ z, float* __restrict__ idx_f, int B) {
  int b = blockIdx.x * (blockDim.x >> 6) + (threadIdx.x >> 6);
  int d = threadIdx.x & 63;
  if (b >= B) return;
  int id = idx[b];
  float zv = cb_f32[(size_t)id * DDIM + d];
  float xv = xn[(size_t)b * DDIM + d];
  z[(size_t)b * DDIM + d]  = zv;
  zq[(size_t)b * DDIM + d] = xv + (zv - xv);   // match reference fp32 rounding exactly
  if (d == 0) idx_f[b] = (float)id;
}

extern "C" void kernel_launch(void* const* d_in, const int* in_sizes, int n_in,
                              void* d_out, int out_size, void* d_ws, size_t ws_size,
                              hipStream_t stream) {
  const float*         x        = (const float*)d_in[0];
  const float*         codebook = (const float*)d_in[1];
  const unsigned char* mask     = (const unsigned char*)d_in[2]; // jnp.bool_ -> 1 byte
  const int*           training = (const int*)d_in[3];

  const int B = in_sizes[0] / DDIM;
  const int K = in_sizes[1] / DDIM;

  // d_out sections: z_q | z | x_n | indices(as float)
  float* out_zq  = (float*)d_out;
  float* out_z   = out_zq + (size_t)B * DDIM;
  float* out_xn  = out_z  + (size_t)B * DDIM;
  float* out_idx = out_xn + (size_t)B * DDIM;

  // workspace layout (256B aligned slices)
  auto align = [](size_t v) { return (v + 255) & ~(size_t)255; };
  char* ws = (char*)d_ws;
  size_t off = 0;
  __bf16* x_hi = (__bf16*)(ws + off); off = align(off + (size_t)B * DDIM * 2);
  __bf16* x_lo = (__bf16*)(ws + off); off = align(off + (size_t)B * DDIM * 2);
  __bf16* c_hi = (__bf16*)(ws + off); off = align(off + (size_t)K * DDIM * 2);
  __bf16* c_lo = (__bf16*)(ws + off); off = align(off + (size_t)K * DDIM * 2);
  float*  c_f  = (float*) (ws + off); off = align(off + (size_t)K * DDIM * 4);
  unsigned short* packM = (unsigned short*)(ws + off);
  off = align(off + (size_t)(B / 16) * K * 2);
  int* idx_i = (int*)(ws + off);

  // 1) normalize x (8 waves/block, 1 wave per row)
  vq_norm_x<<<(B + 7) / 8, 256, 0, stream>>>(x, out_xn, x_hi, x_lo, B);
  // 2) normalize codebook (twice, per reference)
  vq_norm_cb<<<(K + 7) / 8, 256, 0, stream>>>(codebook, c_f, c_hi, c_lo, K);
  // 3) bit-pack mask into per-row-tile u16 words
  {
    size_t total = (size_t)(B / 16) * (size_t)K;
    vq_pack_mask<<<(unsigned)((total + 255) / 256), 256, 0, stream>>>(mask, packM, B, K);
  }
  // 4) fused WMMA distance + masked argmin (1 wave per 16 rows)
  {
    int rowTiles = B / 16;
    vq_argmin_wmma<<<(rowTiles + 7) / 8, 256, 0, stream>>>(x_hi, x_lo, c_hi, c_lo,
                                                           packM, training, idx_i, B, K);
  }
  // 5) gather + finalize outputs
  vq_gather<<<(B + 3) / 4, 256, 0, stream>>>(idx_i, c_f, out_xn, out_zq, out_z, out_idx, B);
}
